// MultiHeadAttention_36395552866460
// MI455X (gfx1250) — matrix operand order repaired and checked by reading, not compiled, not run on hardware
//
#include <hip/hip_runtime.h>
#include <hip/hip_bf16.h>
#include <cstdint>
#include <cstddef>

// ---------------------------------------------------------------------------
// MHA forward for MI455X (gfx1250, wave32, WMMA).
//   x:[2,2048,1024] f32, W_Q/K/V/O:[1024,1024] f32  ->  out:[2,2048,1024] f32
// Pipeline: f32->bf16 convert; bf16 WMMA GEMMs for Q/K/V (V pre-transposed);
// per-wave flash attention (v_wmma_f32_16x16x32_bf16 for QK^T and PV, fp32
// online softmax); bf16 WMMA GEMM for the W_O projection -> fp32 d_out.
// GEMM: wave tile 32x64 (8 WMMA / K-step / wave), A-tile staged through
// GLOBAL_LOAD_ASYNC_TO_LDS_B128 (ASYNCcnt), global_prefetch of next K-tile.
// ---------------------------------------------------------------------------

typedef __bf16 bf16_t;
typedef __attribute__((ext_vector_type(16))) __bf16       v16bf;
typedef __attribute__((ext_vector_type(8)))  float        v8f;
typedef __attribute__((ext_vector_type(4)))  unsigned int v4u;
typedef __attribute__((ext_vector_type(4)))  int          v4i;

#if __has_builtin(__builtin_amdgcn_global_load_async_to_lds_b128) && \
    __has_builtin(__builtin_amdgcn_s_wait_asynccnt)
#define USE_ASYNC_LDS 1
#else
#define USE_ASYNC_LDS 0
#endif

// Builtin signature (from hipcc diagnostics): param0 = v4i __device__* (AS1),
// param1 = v4i __shared__* (AS3), then two integer immediates.
typedef __attribute__((address_space(3))) v4i* lds_v4i_p;
typedef __attribute__((address_space(1))) v4i* gbl_v4i_p;

// Generic->LDS: hardware maps generic LDS-aperture addresses by truncating to
// addr[31:0] (ISA 10.2), so the u32 round-trip yields the LDS offset pointer.
__device__ __forceinline__ lds_v4i_p to_lds(void* p) {
  return (lds_v4i_p)(uintptr_t)(unsigned)(uintptr_t)p;
}
__device__ __forceinline__ gbl_v4i_p to_gbl(const void* p) {
  return (gbl_v4i_p)(uintptr_t)p;
}

union Frag16 {           // 16 bf16 lane-slice of an A or B WMMA operand
  v16bf v;
  v4u   u[2];
};

__device__ __forceinline__ bf16_t f2bf(float f) {
  unsigned int u = __builtin_bit_cast(unsigned int, f);
  u += 0x7FFFu + ((u >> 16) & 1u);          // round-to-nearest-even
  unsigned short h = (unsigned short)(u >> 16);
  return __builtin_bit_cast(bf16_t, h);
}

__device__ __forceinline__ v8f wmma_bf16(v16bf a, v16bf b, v8f c) {
  // (neg_a, A, neg_b, B, c_mod, C, reuse_a, reuse_b)
  return __builtin_amdgcn_wmma_f32_16x16x32_bf16(false, a, false, b,
                                                 (short)0, c, false, false);
}

// ---------------------------------------------------------------------------
// fp32 -> bf16 elementwise conversion
// ---------------------------------------------------------------------------
__global__ void cvt_f32_bf16(const float* __restrict__ in,
                             bf16_t* __restrict__ out, int n) {
  int i = blockIdx.x * blockDim.x + threadIdx.x;
  if (i < n) out[i] = f2bf(in[i]);
}

// ---------------------------------------------------------------------------
// GEMM: C[M,N] = A_bf16[M,K] * B_bf16[K,N], fp32 accum.
// Block tile 128x128, 8 waves (4 M x 2 N), wave tile 32x64, K-step 32.
// MODE 0: bf16 row-major out (Q, K)
// MODE 1: bf16 transposed-per-batch out, Vt[(b*1024 + n)*2048 + s]  (V)
// MODE 2: f32 row-major out (final W_O projection)
// ---------------------------------------------------------------------------
template <int MODE>
__global__ __launch_bounds__(256)
void gemm_kernel(const bf16_t* __restrict__ A, const bf16_t* __restrict__ Bw,
                 void* __restrict__ Cout, int M, int N, int K) {
  constexpr int BM = 128, BN = 128, BK = 32, PAD = 8;
  __shared__ __align__(16) bf16_t As[BM][BK + PAD];    // row-major A tile
  __shared__ __align__(16) bf16_t BsT[BN][BK + PAD];   // transposed B tile

  const int t    = threadIdx.x;
  const int lane = t & 31;
  const int wave = t >> 5;
  const int wm   = wave >> 1;      // 0..3  (M direction, 32 rows each)
  const int wn   = wave & 1;       // 0..1  (N direction, 64 cols each)
  const int half = lane >> 4;      // lane half selects K-subgroup (ISA layout)
  const int lr   = lane & 15;

  const int m0 = blockIdx.y * BM;
  const int n0 = blockIdx.x * BN;

  const int ar = t >> 1;           // A tile: 128 rows, 16 bf16 per thread
  const int ac = (t & 1) * 16;
  const int br = t >> 3;           // B tile: 32 rows, 16 bf16 per thread
  const int bc = (t & 7) * 16;

  v8f acc[2][4] = {};

  for (int k0 = 0; k0 < K; k0 += BK) {
    const bf16_t* asrc = A  + (size_t)(m0 + ar) * K + k0 + ac;
    const bf16_t* bsrc = Bw + (size_t)(k0 + br) * N + n0 + bc;

#if USE_ASYNC_LDS
    // A tile -> LDS via async DMA (2 x 16B per thread), tracked by ASYNCcnt
    __builtin_amdgcn_global_load_async_to_lds_b128(
        to_gbl(asrc),     to_lds(&As[ar][ac]),     0, 0);
    __builtin_amdgcn_global_load_async_to_lds_b128(
        to_gbl(asrc + 8), to_lds(&As[ar][ac + 8]), 0, 0);
#else
    {
      v4u a0 = *(const v4u*)(asrc);
      v4u a1 = *(const v4u*)(asrc + 8);
      *(v4u*)(&As[ar][ac])     = a0;
      *(v4u*)(&As[ar][ac + 8]) = a1;
    }
#endif
    {   // B tile -> LDS transposed (so B-fragments are K-contiguous)
      v4u b0 = *(const v4u*)(bsrc);
      v4u b1 = *(const v4u*)(bsrc + 8);
      const bf16_t* bp0 = (const bf16_t*)&b0;
      const bf16_t* bp1 = (const bf16_t*)&b1;
      #pragma unroll
      for (int j = 0; j < 8; ++j) BsT[bc + j][br]     = bp0[j];
      #pragma unroll
      for (int j = 0; j < 8; ++j) BsT[bc + 8 + j][br] = bp1[j];
    }
    if (k0 + BK < K) {   // pull next K-tile toward L2 (global_prefetch_b8)
      __builtin_prefetch(asrc + (size_t)BK, 0, 1);
      __builtin_prefetch(bsrc + (size_t)BK * N, 0, 1);
    }
#if USE_ASYNC_LDS
    __builtin_amdgcn_s_wait_asynccnt(0);   // our async copies done ...
#endif
    __syncthreads();                        // ... and every wave's visible

    // A fragment: lane lr = row; halves hold K 0..7 / 8..15 then 16..23 / 24..31
    Frag16 af[2], bfr[4];
    #pragma unroll
    for (int mi = 0; mi < 2; ++mi) {
      const bf16_t* p = &As[wm * 32 + mi * 16 + lr][half * 8];
      af[mi].u[0] = *(const v4u*)(p);
      af[mi].u[1] = *(const v4u*)(p + 16);
    }
    // B fragment: lane lr = column; halves hold K 0..7 / 8..15 then 16..23 / 24..31
    #pragma unroll
    for (int ni = 0; ni < 4; ++ni) {
      const bf16_t* p = &BsT[wn * 64 + ni * 16 + lr][half * 8];
      bfr[ni].u[0] = *(const v4u*)(p);
      bfr[ni].u[1] = *(const v4u*)(p + 16);
    }
    #pragma unroll
    for (int mi = 0; mi < 2; ++mi)
      #pragma unroll
      for (int ni = 0; ni < 4; ++ni)
        acc[mi][ni] = wmma_bf16(af[mi].v, bfr[ni].v, acc[mi][ni]);

    __syncthreads();
  }

  // Epilogue. C/D layout: VGPR r holds row r (lanes 0-15) / row r+8 (16-31),
  // column = lane%16.
  #pragma unroll
  for (int mi = 0; mi < 2; ++mi) {
    #pragma unroll
    for (int ni = 0; ni < 4; ++ni) {
      const int col = n0 + wn * 64 + ni * 16 + lr;
      #pragma unroll
      for (int r = 0; r < 8; ++r) {
        const int m = m0 + wm * 32 + mi * 16 + r + 8 * half;
        const float val = acc[mi][ni][r];
        if (MODE == 0) {
          ((bf16_t*)Cout)[(size_t)m * N + col] = f2bf(val);
        } else if (MODE == 1) {
          const int bb = m >> 11;            // batch = m / 2048
          const int s  = m & 2047;
          ((bf16_t*)Cout)[((size_t)(bb * 1024 + col)) * 2048 + s] = f2bf(val);
        } else {
          ((float*)Cout)[(size_t)m * N + col] = val;
        }
      }
    }
  }
}

// ---------------------------------------------------------------------------
// Flash attention: one wave per (b, h, 16-query tile). 32 keys per step:
// 4 WMMAs for S = Q K^T, fp32 online softmax (shfl reductions inside the
// 16-lane halves matching the C/D layout), P restaged via LDS into A-layout,
// 4 WMMAs for O += P V using V pre-transposed to [b][h*64+d][s].
// ---------------------------------------------------------------------------
__global__ __launch_bounds__(32)
void attn_kernel(const bf16_t* __restrict__ Q, const bf16_t* __restrict__ Kmat,
                 const bf16_t* __restrict__ Vt, bf16_t* __restrict__ AO) {
  __shared__ __align__(16) bf16_t Pt[16][32];

  const int lane = threadIdx.x & 31;
  const int half = lane >> 4;
  const int lr   = lane & 15;
  const int qt   = blockIdx.x;          // 0..127 query tile
  const int bh   = blockIdx.y;          // 0..31
  const int bb   = bh >> 4;
  const int h    = bh & 15;
  const int q0   = qt * 16;
  const float scale = 0.125f;           // 1/sqrt(64)
  const float NEG = -3.0e38f;

  // Q fragments (A-layout), K-dim 64 split into two 32-chunks
  Frag16 qf[2];
  {
    const bf16_t* qrow = Q + (size_t)(bb * 2048 + q0 + lr) * 1024 + h * 64;
    #pragma unroll
    for (int c = 0; c < 2; ++c) {
      const bf16_t* p = qrow + c * 32 + half * 8;
      qf[c].u[0] = *(const v4u*)(p);
      qf[c].u[1] = *(const v4u*)(p + 16);
    }
  }

  v8f o[4] = {};
  float mrow[8], lrow[8];
  #pragma unroll
  for (int r = 0; r < 8; ++r) { mrow[r] = NEG; lrow[r] = 0.0f; }

  const int kend = q0 + 15;
  for (int kv = 0; kv <= kend; kv += 32) {
    // --- scores S[16q x 32k]: two 16-column tiles, 2 WMMAs each over d=64
    v8f s[2];
    #pragma unroll
    for (int nt = 0; nt < 2; ++nt) {
      const int key = kv + nt * 16 + lr;          // lane lr = key column
      const bf16_t* krow = Kmat + (size_t)(bb * 2048 + key) * 1024 + h * 64;
      Frag16 kf0, kf1;                            // B = K^T: K-contiguous in d
      kf0.u[0] = *(const v4u*)(krow + half * 8);
      kf0.u[1] = *(const v4u*)(krow + half * 8 + 16);
      kf1.u[0] = *(const v4u*)(krow + 32 + half * 8);
      kf1.u[1] = *(const v4u*)(krow + 32 + half * 8 + 16);
      v8f a = {};
      a = wmma_bf16(qf[0].v, kf0.v, a);
      a = wmma_bf16(qf[1].v, kf1.v, a);
      s[nt] = a;
    }

    // --- online softmax over the 32 new columns
    #pragma unroll
    for (int r = 0; r < 8; ++r) {
      const int qi = q0 + r + 8 * half;           // this lane-half's row
      float v0 = s[0][r] * scale;
      float v1 = s[1][r] * scale;
      if (kv + lr > qi)      v0 = NEG;            // causal mask
      if (kv + 16 + lr > qi) v1 = NEG;
      float m = fmaxf(v0, v1);
      #pragma unroll
      for (int d = 1; d < 16; d <<= 1) m = fmaxf(m, __shfl_xor(m, d, 32));
      const float mn    = fmaxf(mrow[r], m);
      const float alpha = __expf(mrow[r] - mn);
      const float p0    = __expf(v0 - mn);
      const float p1    = __expf(v1 - mn);
      float ls = p0 + p1;
      #pragma unroll
      for (int d = 1; d < 16; d <<= 1) ls += __shfl_xor(ls, d, 32);
      lrow[r] = lrow[r] * alpha + ls;
      mrow[r] = mn;
      #pragma unroll
      for (int dt = 0; dt < 4; ++dt) o[dt][r] *= alpha;
      Pt[r + 8 * half][lr]      = f2bf(p0);       // stage P (D-layout order)
      Pt[r + 8 * half][16 + lr] = f2bf(p1);
    }
    __syncthreads();

    // --- O[16 x 64] += P[16 x 32] * V[32 x 64]
    Frag16 pf;                                    // reload P in A-layout
    {
      const bf16_t* pp = &Pt[lr][half * 8];
      pf.u[0] = *(const v4u*)(pp);
      pf.u[1] = *(const v4u*)(pp + 16);
    }
    #pragma unroll
    for (int dt = 0; dt < 4; ++dt) {
      Frag16 vf;                                  // B from transposed V: K-contig
      const bf16_t* vp = Vt +
          ((size_t)(bb * 1024 + h * 64 + dt * 16 + lr)) * 2048 + kv + half * 8;
      vf.u[0] = *(const v4u*)(vp);
      vf.u[1] = *(const v4u*)(vp + 16);
      o[dt] = wmma_bf16(pf.v, vf.v, o[dt]);
    }
    __syncthreads();
  }

  // --- normalize and store attn output (bf16, row-major [b*s][h*64+d])
  #pragma unroll
  for (int r = 0; r < 8; ++r) {
    const float inv = 1.0f / lrow[r];
    const size_t orow = (size_t)(bb * 2048 + q0 + r + 8 * half) * 1024 + h * 64;
    #pragma unroll
    for (int dt = 0; dt < 4; ++dt)
      AO[orow + dt * 16 + lr] = f2bf(o[dt][r] * inv);
  }
}

// ---------------------------------------------------------------------------
extern "C" void kernel_launch(void* const* d_in, const int* in_sizes, int n_in,
                              void* d_out, int out_size, void* d_ws, size_t ws_size,
                              hipStream_t stream) {
  (void)in_sizes; (void)n_in; (void)out_size; (void)ws_size;
  const float* x  = (const float*)d_in[0];
  const float* Wq = (const float*)d_in[1];
  const float* Wk = (const float*)d_in[2];
  const float* Wv = (const float*)d_in[3];
  const float* Wo = (const float*)d_in[4];
  float* out = (float*)d_out;

  constexpr int BS = 2, SEQ = 2048, DM = 1024;
  const size_t rows = (size_t)BS * SEQ;           // 4096
  const size_t xN   = rows * DM;                  // 4,194,304
  const size_t wN   = (size_t)DM * DM;            // 1,048,576

  // Workspace carve-up (bf16): x, 4 weights, Q, K, Vt, attn_out  (~48 MB)
  char* w = (char*)d_ws;
  bf16_t* xb  = (bf16_t*)w; w += xN * sizeof(bf16_t);
  bf16_t* wqb = (bf16_t*)w; w += wN * sizeof(bf16_t);
  bf16_t* wkb = (bf16_t*)w; w += wN * sizeof(bf16_t);
  bf16_t* wvb = (bf16_t*)w; w += wN * sizeof(bf16_t);
  bf16_t* wob = (bf16_t*)w; w += wN * sizeof(bf16_t);
  bf16_t* Qb  = (bf16_t*)w; w += xN * sizeof(bf16_t);
  bf16_t* Kb  = (bf16_t*)w; w += xN * sizeof(bf16_t);
  bf16_t* Vtb = (bf16_t*)w; w += xN * sizeof(bf16_t);
  bf16_t* AOb = (bf16_t*)w; w += xN * sizeof(bf16_t);

  // 1) fp32 -> bf16
  cvt_f32_bf16<<<(int)(xN / 256), 256, 0, stream>>>(x,  xb,  (int)xN);
  cvt_f32_bf16<<<(int)(wN / 256), 256, 0, stream>>>(Wq, wqb, (int)wN);
  cvt_f32_bf16<<<(int)(wN / 256), 256, 0, stream>>>(Wk, wkb, (int)wN);
  cvt_f32_bf16<<<(int)(wN / 256), 256, 0, stream>>>(Wv, wvb, (int)wN);
  cvt_f32_bf16<<<(int)(wN / 256), 256, 0, stream>>>(Wo, wob, (int)wN);

  // 2) projections (WMMA GEMMs). grid = (N/128, M/128)
  dim3 gblk(256);
  dim3 ggrd(DM / 128, (unsigned)(rows / 128));    // (8, 32)
  gemm_kernel<0><<<ggrd, gblk, 0, stream>>>(xb, wqb, Qb,  (int)rows, DM, DM);
  gemm_kernel<0><<<ggrd, gblk, 0, stream>>>(xb, wkb, Kb,  (int)rows, DM, DM);
  gemm_kernel<1><<<ggrd, gblk, 0, stream>>>(xb, wvb, Vtb, (int)rows, DM, DM);

  // 3) flash attention: one wave per (q-tile, b*h)
  attn_kernel<<<dim3(SEQ / 16, BS * 16), dim3(32), 0, stream>>>(Qb, Kb, Vtb, AOb);

  // 4) output projection -> fp32 d_out
  gemm_kernel<2><<<ggrd, gblk, 0, stream>>>(AOb, wob, out, (int)rows, DM, DM);
}
